// HashGrid_w_pose_54365696033029
// MI455X (gfx1250) — matrix operand order repaired and checked by reading, not compiled, not run on hardware
//
#include <hip/hip_runtime.h>
#include <stdint.h>

// instant-NGP config matching the reference
#define NLVL   16
#define TBITS  19
#define TMASK  ((1u << TBITS) - 1u)
#define PRIME1 2654435761u
#define PRIME2 805459861u
#define PRIME3 3674653429u

typedef _Float16 v16h __attribute__((ext_vector_type(16)));
typedef _Float16 v8h  __attribute__((ext_vector_type(8)));
typedef _Float16 v2h  __attribute__((ext_vector_type(2)));
typedef float    v8f  __attribute__((ext_vector_type(8)));

// floor(16 * 1.5^l), l = 0..15 (matches np.floor in fp64 then fp32 cast)
__constant__ float c_res[NLVL] = {
    16.f, 24.f, 36.f, 54.f, 81.f, 121.f, 182.f, 273.f,
    410.f, 615.f, 922.f, 1383.f, 2075.f, 3113.f, 4670.f, 7006.f};

// ---- hash-grid level encodings (fp32 math, tables are [L][2^19][2] f32) ----
__device__ __forceinline__ float2 enc_d3(float x, float y, float z, int lvl,
                                         const float2* __restrict__ tab) {
    float r  = c_res[lvl];
    float px = x * r, py = y * r, pz = z * r;
    float fx = floorf(px), fy = floorf(py), fz = floorf(pz);
    float wx = px - fx, wy = py - fy, wz = pz - fz;
    float ox = 1.f - wx, oy = 1.f - wy, oz = 1.f - wz;
    uint32_t ux = (uint32_t)fx, uy = (uint32_t)fy, uz = (uint32_t)fz;
    uint32_t hy0 = uy * PRIME1, hy1 = hy0 + PRIME1;
    uint32_t hz0 = uz * PRIME2, hz1 = hz0 + PRIME2;
    uint32_t bi  = (uint32_t)lvl << TBITS;
    float f0 = 0.f, f1 = 0.f;
#pragma unroll
    for (int cc = 0; cc < 8; ++cc) {
        uint32_t h = (ux + (cc & 1)) ^ ((cc & 2) ? hy1 : hy0) ^ ((cc & 4) ? hz1 : hz0);
        float w = ((cc & 1) ? wx : ox) * ((cc & 2) ? wy : oy) * ((cc & 4) ? wz : oz);
        float2 t = tab[bi + (h & TMASK)];
        f0 = fmaf(w, t.x, f0);
        f1 = fmaf(w, t.y, f1);
    }
    return make_float2(f0, f1);
}

__device__ __forceinline__ float2 enc_d4(float x, float y, float z, float q, int lvl,
                                         const float2* __restrict__ tab) {
    float r  = c_res[lvl];
    float px = x * r, py = y * r, pz = z * r, pq = q * r;
    float fx = floorf(px), fy = floorf(py), fz = floorf(pz), fq = floorf(pq);
    float wx = px - fx, wy = py - fy, wz = pz - fz, wq = pq - fq;
    float ox = 1.f - wx, oy = 1.f - wy, oz = 1.f - wz, oq = 1.f - wq;
    uint32_t ux = (uint32_t)fx, uy = (uint32_t)fy, uz = (uint32_t)fz, uq = (uint32_t)fq;
    uint32_t hy0 = uy * PRIME1, hy1 = hy0 + PRIME1;
    uint32_t hz0 = uz * PRIME2, hz1 = hz0 + PRIME2;
    uint32_t hq0 = uq * PRIME3, hq1 = hq0 + PRIME3;
    uint32_t bi  = (uint32_t)lvl << TBITS;
    float f0 = 0.f, f1 = 0.f;
#pragma unroll
    for (int cc = 0; cc < 16; ++cc) {
        uint32_t h = (ux + (cc & 1)) ^ ((cc & 2) ? hy1 : hy0) ^
                     ((cc & 4) ? hz1 : hz0) ^ ((cc & 8) ? hq1 : hq0);
        float w = ((cc & 1) ? wx : ox) * ((cc & 2) ? wy : oy) *
                  ((cc & 4) ? wz : oz) * ((cc & 8) ? wq : oq);
        float2 t = tab[bi + (h & TMASK)];
        f0 = fmaf(w, t.x, f0);
        f1 = fmaf(w, t.y, f1);
    }
    return make_float2(f0, f1);
}

// ---- WMMA operand loaders (ISA 16-bit A / B VGPR layouts, wave32) ----
// A 16x32 f16: lanes 0-15 hold row m=lane, K {kb..kb+7, kb+16..kb+23};
//              lanes 16-31 hold row m=lane-16, K {kb+8..kb+15, kb+24..kb+31}.
__device__ __forceinline__ v16h load_a(const _Float16* buf, int stride, int kb, int lane) {
    int m = lane & 15;
    int o = kb + ((lane >> 4) << 3);
    const _Float16* p = buf + m * stride + o;
    v8h lo = *(const v8h*)p;
    v8h hi = *(const v8h*)(p + 16);
    return __builtin_shufflevector(lo, hi, 0, 1, 2, 3, 4, 5, 6, 7,
                                   8, 9, 10, 11, 12, 13, 14, 15);
}

// B 32x16 f16 (weights stored column-major [n][kdim] in LDS):
// lanes 0-15 hold col n=nb+lane, K {kb..kb+7, kb+16..kb+23}; lanes 16-31 hold K {kb+8..kb+15, kb+24..kb+31}.
__device__ __forceinline__ v16h load_b(const _Float16* wt, int kdim, int nb, int kb, int lane) {
    int nn = nb + (lane & 15);
    int o  = kb + ((lane >> 4) << 3);
    const _Float16* p = wt + nn * kdim + o;
    v8h lo = *(const v8h*)p;
    v8h hi = *(const v8h*)(p + 16);
    return __builtin_shufflevector(lo, hi, 0, 1, 2, 3, 4, 5, 6, 7,
                                   8, 9, 10, 11, 12, 13, 14, 15);
}

__global__ __launch_bounds__(256) void ngp_pose_fused(
    const float* __restrict__ fc, const float* __restrict__ fn,
    const float* __restrict__ pe,
    const float* __restrict__ ptab, const float* __restrict__ ntab,
    const float* __restrict__ qtab,
    const float* __restrict__ W1, const float* __restrict__ W2,
    const float* __restrict__ W3,
    float* __restrict__ out, int n) {
    // 62 KB static LDS: f16 weights (col-major) + per-wave enc / hidden tiles
    __shared__ __align__(16) _Float16 sW1[64 * 96];       // 12 KB
    __shared__ __align__(16) _Float16 sW2[64 * 64];       // 8 KB
    __shared__ __align__(16) _Float16 sW3[16 * 64];       // 2 KB (cols 9..15 zero)
    __shared__ __align__(16) _Float16 sEnc[8][16 * 96];   // 24 KB (reused as hid2)
    __shared__ __align__(16) _Float16 sHid[8][16 * 64];   // 16 KB

    const int tid  = threadIdx.x;
    const int wv   = tid >> 5;
    const int lane = tid & 31;

    // --- stage weights into LDS as f16, transposed to column-major ---
    for (int i = tid; i < 96 * 64; i += 256) {
        int k = i / 64, nn = i % 64;
        sW1[nn * 96 + k] = (_Float16)W1[i];
    }
    for (int i = tid; i < 64 * 64; i += 256) {
        int k = i / 64, nn = i % 64;
        sW2[nn * 64 + k] = (_Float16)W2[i];
    }
    for (int i = tid; i < 16 * 64; i += 256) {
        int nn = i / 64, k = i % 64;
        sW3[nn * 64 + k] = (nn < 9) ? (_Float16)W3[k * 9 + nn] : (_Float16)0.f;
    }
    __syncthreads();

    const int base    = blockIdx.x * 128 + wv * 16;  // 16 points per wave
    _Float16* enc     = &sEnc[wv][0];                // [16 rows][96 f16]
    _Float16* hid     = &sHid[wv][0];                // [16 rows][64 f16]
    _Float16* hid2    = enc;                         // enc dead after A-tiles read

    // --- encoding: each lane owns one (point, level); 2 points x 16 levels / iter ---
#pragma unroll 1
    for (int it = 0; it < 8; ++it) {
        int pt  = it * 2 + (lane >> 4);
        int lvl = lane & 15;
        int g   = base + pt;
        if (g >= n) g = n - 1;  // clamp tail (stores are predicated later)

        float x = fc[g * 3 + 0], y = fc[g * 3 + 1], z = fc[g * 3 + 2];
        float2 e = enc_d3(x, y, z, lvl, (const float2*)ptab);
        *(v2h*)(enc + pt * 96 + lvl * 2) = (v2h){(_Float16)e.x, (_Float16)e.y};

        x = fn[g * 3 + 0]; y = fn[g * 3 + 1]; z = fn[g * 3 + 2];
        e = enc_d3(x, y, z, lvl, (const float2*)ntab);
        *(v2h*)(enc + pt * 96 + 32 + lvl * 2) = (v2h){(_Float16)e.x, (_Float16)e.y};

        x = pe[g * 4 + 0]; y = pe[g * 4 + 1]; z = pe[g * 4 + 2];
        float q = pe[g * 4 + 3];
        e = enc_d4(x, y, z, q, lvl, (const float2*)qtab);
        *(v2h*)(enc + pt * 96 + 64 + lvl * 2) = (v2h){(_Float16)e.x, (_Float16)e.y};
    }
    // wave-private enc buffer: DS pipe is in-order per wave; pin HW+compiler order
    asm volatile("s_wait_dscnt 0" ::: "memory");

    const int col = lane & 15;
    const int mb  = (lane >> 4) * 8;

    // --- layer 1: [16x96] @ [96x64], K in 3 chunks of 32 ---
    v16h a0 = load_a(enc, 96, 0, lane);
    v16h a1 = load_a(enc, 96, 32, lane);
    v16h a2 = load_a(enc, 96, 64, lane);
#pragma unroll
    for (int nt = 0; nt < 4; ++nt) {
        v8f acc = {};
        acc = __builtin_amdgcn_wmma_f32_16x16x32_f16(false, a0, false,
                  load_b(sW1, 96, nt * 16, 0, lane), (short)0, acc, false, false);
        acc = __builtin_amdgcn_wmma_f32_16x16x32_f16(false, a1, false,
                  load_b(sW1, 96, nt * 16, 32, lane), (short)0, acc, false, false);
        acc = __builtin_amdgcn_wmma_f32_16x16x32_f16(false, a2, false,
                  load_b(sW1, 96, nt * 16, 64, lane), (short)0, acc, false, false);
#pragma unroll
        for (int rr = 0; rr < 8; ++rr) {
            float v = acc[rr];
            v = v > 0.f ? v : 0.f;  // ReLU
            hid[(mb + rr) * 64 + nt * 16 + col] = (_Float16)v;
        }
    }
    asm volatile("s_wait_dscnt 0" ::: "memory");

    // --- layer 2: [16x64] @ [64x64] ---
    v16h h0 = load_a(hid, 64, 0, lane);
    v16h h1 = load_a(hid, 64, 32, lane);
#pragma unroll
    for (int nt = 0; nt < 4; ++nt) {
        v8f acc = {};
        acc = __builtin_amdgcn_wmma_f32_16x16x32_f16(false, h0, false,
                  load_b(sW2, 64, nt * 16, 0, lane), (short)0, acc, false, false);
        acc = __builtin_amdgcn_wmma_f32_16x16x32_f16(false, h1, false,
                  load_b(sW2, 64, nt * 16, 32, lane), (short)0, acc, false, false);
#pragma unroll
        for (int rr = 0; rr < 8; ++rr) {
            float v = acc[rr];
            v = v > 0.f ? v : 0.f;  // ReLU
            hid2[(mb + rr) * 64 + nt * 16 + col] = (_Float16)v;
        }
    }
    asm volatile("s_wait_dscnt 0" ::: "memory");

    // --- layer 3: [16x64] @ [64x9] (padded to 16 cols) ---
    v16h g0 = load_a(hid2, 64, 0, lane);
    v16h g1 = load_a(hid2, 64, 32, lane);
    v8f acc = {};
    acc = __builtin_amdgcn_wmma_f32_16x16x32_f16(false, g0, false,
              load_b(sW3, 64, 0, 0, lane), (short)0, acc, false, false);
    acc = __builtin_amdgcn_wmma_f32_16x16x32_f16(false, g1, false,
              load_b(sW3, 64, 0, 32, lane), (short)0, acc, false, false);

    if (col < 9) {
#pragma unroll
        for (int rr = 0; rr < 8; ++rr) {
            int g = base + mb + rr;
            if (g < n) out[g * 9 + col] = acc[rr];
        }
    }
}

extern "C" void kernel_launch(void* const* d_in, const int* in_sizes, int n_in,
                              void* d_out, int out_size, void* d_ws, size_t ws_size,
                              hipStream_t stream) {
    const float* fc   = (const float*)d_in[0];  // face_centers  [N,3]
    const float* fn   = (const float*)d_in[1];  // face_normals  [N,3]
    const float* pe   = (const float*)d_in[2];  // pose_extended [N,4]
    const float* ptab = (const float*)d_in[3];  // pos_table     [16,2^19,2]
    const float* ntab = (const float*)d_in[4];  // normal_table
    const float* qtab = (const float*)d_in[5];  // pose_table
    const float* W1   = (const float*)d_in[6];  // [96,64]
    const float* W2   = (const float*)d_in[7];  // [64,64]
    const float* W3   = (const float*)d_in[8];  // [64,9]
    float*       out  = (float*)d_out;          // [N,9]

    int n = in_sizes[0] / 3;
    int blocks = (n + 127) / 128;  // 128 points per 256-thread (8-wave) block
    ngp_pose_fused<<<blocks, 256, 0, stream>>>(fc, fn, pe, ptab, ntab, qtab,
                                               W1, W2, W3, out, n);
}
